// LSTM_73177652789693
// MI455X (gfx1250) — matrix operand order repaired and checked by reading, not compiled, not run on hardware
//
#include <hip/hip_runtime.h>
#include <hip/hip_bf16.h>

// ---------------------------------------------------------------------------
// Stacked LSTM: L=8, B=256, T=8, E=H=1024.
// Strategy:
//   1) pack_weights: convert all gate weights f32 -> bf16 once, laid out in
//      lstm_step's B-fragment order (lane = column, K-consecutive per half wave).
//      134 MB bf16 -> resident in MI455X's 192 MB L2 across all 64 steps.
//   2) 64x lstm_step kernels (one per (t,l)): fused 4-gate GEMM via
//      v_wmma_f32_16x16x32_bf16 + bias + activations + cell update.
// ---------------------------------------------------------------------------

typedef __attribute__((ext_vector_type(16))) __bf16 v16bf;
typedef __attribute__((ext_vector_type(8)))  float  v8f;

union ABFrag {
    uint4 u[2];
    v16bf v;
};

#define LSTM_L 8
#define LSTM_B 256
#define LSTM_T 8
#define LSTM_E 1024
#define LSTM_H 1024
#define LSTM_K 2048          // E + H
#define KC     256           // K-chunk staged in LDS
#define SA     264           // padded LDS row stride (bf16 elems), mult of 8

__device__ __forceinline__ unsigned int pack_bf16x2(float a, float b) {
    unsigned int ua = __builtin_bit_cast(unsigned int, a);
    unsigned int ub = __builtin_bit_cast(unsigned int, b);
    ua = (ua + 0x7FFFu + ((ua >> 16) & 1u)) >> 16;   // round-to-nearest-even
    ub = (ub + 0x7FFFu + ((ub >> 16) & 1u)) >> 16;
    return ua | (ub << 16);
}

__device__ __forceinline__ float fast_sigmoid(float x) {
    return 1.0f / (1.0f + __expf(-x));
}

// ---------------------------------------------------------------------------
// Weight packing: pw layout, per (layer l, gate g):
//   [kblk 0..63][nblk 0..63][lane 0..31][8 dwords]  (512 bf16 per 32x16 block)
// lane (0..15): column n = nblk*16+lane, K = kblk*32 + 0..15 (2 per dword)
// lane (16..31): same column, K = kblk*32 + 16..31
// One thread packs one dword (2 bf16). Total dwords = 32*64*64*256 = 33.5M.
// ---------------------------------------------------------------------------
__global__ __launch_bounds__(256)
void lstm_pack_weights(const float* __restrict__ Wu, const float* __restrict__ Wf,
                       const float* __restrict__ Wo, const float* __restrict__ Wc,
                       unsigned int* __restrict__ pw32) {
    size_t d = (size_t)blockIdx.x * 256 + threadIdx.x;
    int dword = (int)(d & 255);
    int lane  = dword >> 3;
    int dw    = dword & 7;
    size_t blk = d >> 8;
    int nblk = (int)(blk & 63);
    int kblk = (int)((blk >> 6) & 63);
    int lg   = (int)(blk >> 12);        // 0..31
    int g    = lg & 3;
    int l    = lg >> 2;

    const float* W = (g == 0) ? Wu : (g == 1) ? Wf : (g == 2) ? Wo : Wc;

    int n = nblk * 16 + (lane & 15);
    int k = kblk * 32 + ((lane >> 4) << 4) + dw * 2;
    size_t base = (size_t)l * LSTM_K * LSTM_H + (size_t)k * LSTM_H + n;
    float a = W[base];
    float b = W[base + LSTM_H];
    pw32[d] = pack_bf16x2(a, b);
}

// ---------------------------------------------------------------------------
// Fused LSTM step for one (t, layer).
// grid = (16 mtiles, 16 ntiles), block = 128 threads (4 waves, wave g = gate g)
// Each wave: 16 rows x 64 cols f32 accumulation over K=2048 via WMMA bf16.
// ---------------------------------------------------------------------------
__global__ __launch_bounds__(128)
void lstm_step(const float* __restrict__ inp, long inp_stride,  // B x 1024
               const float* __restrict__ hrec,                  // B x H (t-1)
               const unsigned short* __restrict__ pw,           // packed, this layer (4 gates)
               const float* __restrict__ bu, const float* __restrict__ bfg,
               const float* __restrict__ bo, const float* __restrict__ bc,
               float* __restrict__ c_state,                     // B x H, in-place
               float* __restrict__ h_out,                       // B x H (t)
               float* __restrict__ out_hid,                     // B x T x H or null
               float* __restrict__ out_mem,                     // B x T x H or null
               float* __restrict__ out_ht,                      // B x H or null
               float* __restrict__ out_ct,                      // B x H or null
               int t) {
    __shared__ unsigned short As[16 * SA];        // bf16 A chunk: 16 rows x KC
    __shared__ float gbuf[4 * 16 * 65];           // per-gate 16x64 staging

    const int tid  = threadIdx.x;
    const int wave = tid >> 5;
    const int lane = tid & 31;
    const int r0 = blockIdx.x * 16;               // batch rows
    const int n0 = blockIdx.y * 64;               // output columns

    v8f acc[4];
    for (int j = 0; j < 4; ++j) acc[j] = (v8f){0.f,0.f,0.f,0.f,0.f,0.f,0.f,0.f};

    const unsigned short* pwg = pw + (size_t)wave * (64u * 64u * 512u);

    for (int kc = 0; kc < LSTM_K; kc += KC) {
        const float* src;
        long sstride;
        int kofs;
        if (kc < LSTM_E) { src = inp;  sstride = inp_stride; kofs = kc; }
        else             { src = hrec; sstride = LSTM_H;     kofs = kc - LSTM_E; }

        __syncthreads();   // previous chunk's ds_loads complete
        // cooperative load of 16 x 256 f32, convert -> bf16 into LDS
        #pragma unroll
        for (int i = 0; i < 8; ++i) {
            int idx4 = tid + i * 128;             // 0..1023 float4s
            int row  = idx4 >> 6;                 // 64 float4 per row
            int kk4  = idx4 & 63;
            const float4 v = *(const float4*)&src[(long)(r0 + row) * sstride + kofs + kk4 * 4];
            uint2 pk;
            pk.x = pack_bf16x2(v.x, v.y);
            pk.y = pack_bf16x2(v.z, v.w);
            *(uint2*)&As[row * SA + kk4 * 4] = pk;
        }
        __syncthreads();

        const int arow  = lane & 15;
        const int khalf = (lane >> 4) << 4;
        #pragma unroll
        for (int ks = 0; ks < KC; ks += 32) {
            const int kblk = (kc + ks) >> 5;
            ABFrag a;
            const int abase = arow * SA + ks + khalf;
            a.u[0] = *(const uint4*)&As[abase];        // K 0..7   (of this 32-blk, +16 for hi lanes)
            a.u[1] = *(const uint4*)&As[abase + 8];    // K 8..15  (+16 for hi lanes)
            #pragma unroll
            for (int j = 0; j < 4; ++j) {
                ABFrag b;
                const unsigned short* p =
                    pwg + (((size_t)kblk * 64 + (blockIdx.y * 4 + j)) << 9) + (lane << 4);
                b.u[0] = *(const uint4*)p;
                b.u[1] = *(const uint4*)(p + 8);
                acc[j] = __builtin_amdgcn_wmma_f32_16x16x32_bf16(
                    false, a.v, false, b.v, (short)0, acc[j], false, false);
            }
        }
    }

    // bias + activation, stage gates to LDS
    const float* bias = (wave == 0) ? bu : (wave == 1) ? bfg : (wave == 2) ? bo : bc;
    const int mbase = (lane >> 4) << 3;
    #pragma unroll
    for (int j = 0; j < 4; ++j) {
        const int nl = j * 16 + (lane & 15);
        const float bval = bias[n0 + nl];
        #pragma unroll
        for (int i = 0; i < 8; ++i) {
            float v = acc[j][i] + bval;
            v = (wave < 3) ? fast_sigmoid(v) : tanhf(v);
            gbuf[wave * 1040 + (mbase + i) * 65 + nl] = v;
        }
    }
    __syncthreads();

    // cell update: 16 x 64 elements, 128 threads x 8
    #pragma unroll
    for (int i = 0; i < 8; ++i) {
        const int e  = tid + i * 128;
        const int m  = e >> 6;
        const int nl = e & 63;
        const int gi = m * 65 + nl;
        const float u  = gbuf[0 * 1040 + gi];
        const float f  = gbuf[1 * 1040 + gi];
        const float o  = gbuf[2 * 1040 + gi];
        const float ct = gbuf[3 * 1040 + gi];
        const long gidx = (long)(r0 + m) * LSTM_H + (n0 + nl);
        const float cold = c_state[gidx];
        const float cnew = f * cold + u * ct;
        const float hnew = o * tanhf(cnew);
        c_state[gidx] = cnew;
        h_out[gidx]   = hnew;
        if (out_hid) {
            const long oidx = (long)(r0 + m) * (LSTM_T * LSTM_H) + (long)t * LSTM_H + (n0 + nl);
            out_hid[oidx] = hnew;
            out_mem[oidx] = cnew;
            if (out_ht) { out_ht[gidx] = hnew; out_ct[gidx] = cnew; }
        }
    }
}

// ---------------------------------------------------------------------------
extern "C" void kernel_launch(void* const* d_in, const int* in_sizes, int n_in,
                              void* d_out, int out_size, void* d_ws, size_t ws_size,
                              hipStream_t stream) {
    (void)in_sizes; (void)n_in; (void)out_size; (void)ws_size;

    const float* x  = (const float*)d_in[0];   // (B,T,E)
    const float* h0 = (const float*)d_in[1];   // (L,B,H)
    const float* c0 = (const float*)d_in[2];   // (L,B,H)
    const float* Wu = (const float*)d_in[3];   // (L,2048,1024)
    const float* bu = (const float*)d_in[4];
    const float* Wf = (const float*)d_in[5];
    const float* bf = (const float*)d_in[6];
    const float* Wo = (const float*)d_in[7];
    const float* bo = (const float*)d_in[8];
    const float* Wc = (const float*)d_in[9];
    const float* bc = (const float*)d_in[10];

    float* out_hid = (float*)d_out;                                        // (B,T,H)
    float* out_mem = out_hid + (size_t)LSTM_B * LSTM_T * LSTM_H;           // (B,T,H)
    float* out_ht  = out_mem + (size_t)LSTM_B * LSTM_T * LSTM_H;           // (B,H)
    float* out_ct  = out_ht  + (size_t)LSTM_B * LSTM_H;                    // (B,H)

    // workspace layout
    const size_t PW_BYTES = (size_t)4 * LSTM_L * LSTM_K * LSTM_H * 2;      // 134 MB bf16
    const size_t STATE_ELEMS = (size_t)LSTM_L * LSTM_B * LSTM_H;           // 2M floats
    char* wsb = (char*)d_ws;
    unsigned short* pw = (unsigned short*)wsb;
    float* hA = (float*)(wsb + PW_BYTES);
    float* hB = hA + STATE_ELEMS;
    float* cS = hB + STATE_ELEMS;

    // 1) pack weights f32 -> bf16 in WMMA fragment order (L2-resident afterwards)
    {
        const size_t ndwords = PW_BYTES / 4;                               // 33.5M
        lstm_pack_weights<<<dim3((unsigned)(ndwords / 256)), 256, 0, stream>>>(
            Wu, Wf, Wo, Wc, (unsigned int*)pw);
    }

    // 2) initial states into workspace (h double-buffered, c in-place)
    hipMemcpyAsync(hA, h0, STATE_ELEMS * sizeof(float), hipMemcpyDeviceToDevice, stream);
    hipMemcpyAsync(cS, c0, STATE_ELEMS * sizeof(float), hipMemcpyDeviceToDevice, stream);

    // 3) 64 sequential fused steps
    float* hbuf[2] = {hA, hB};
    for (int t = 0; t < LSTM_T; ++t) {
        float* hprev_all = hbuf[t & 1];          // h at time t-1 (t=0: h0)
        float* hcur_all  = hbuf[(t & 1) ^ 1];    // h at time t (being written)
        for (int l = 0; l < LSTM_L; ++l) {
            const float* inp;
            long istride;
            if (l == 0) { inp = x + (size_t)t * LSTM_E; istride = (long)LSTM_T * LSTM_E; }
            else        { inp = hcur_all + (size_t)(l - 1) * LSTM_B * LSTM_H; istride = LSTM_H; }
            const unsigned short* pwl = pw + (size_t)(l * 4) * 64u * 64u * 512u;
            const bool last = (l == LSTM_L - 1);
            lstm_step<<<dim3(16, 16), 128, 0, stream>>>(
                inp, istride,
                hprev_all + (size_t)l * LSTM_B * LSTM_H,
                pwl,
                bu + (size_t)l * LSTM_H, bf + (size_t)l * LSTM_H,
                bo + (size_t)l * LSTM_H, bc + (size_t)l * LSTM_H,
                cS + (size_t)l * LSTM_B * LSTM_H,
                hcur_all + (size_t)l * LSTM_B * LSTM_H,
                last ? out_hid : nullptr,
                last ? out_mem : nullptr,
                (last && t == LSTM_T - 1) ? out_ht : nullptr,
                (last && t == LSTM_T - 1) ? out_ct : nullptr,
                t);
        }
    }
}
